// ImprovedLSTMPredictor_52475910422674
// MI455X (gfx1250) — matrix operand order repaired and checked by reading, not compiled, not run on hardware
//
#include <hip/hip_runtime.h>
#include <math.h>
#include <stdint.h>

#define B_ 256
#define S_ 128
#define T_ 64
#define IN_ 4
#define MID_ 128
#define HID_ 512
#define NH_ 4

typedef _Float16 h16v __attribute__((ext_vector_type(16)));
typedef _Float16 h8v  __attribute__((ext_vector_type(8)));
typedef float    f8v  __attribute__((ext_vector_type(8)));

// ---------------------------------------------------------------------------
// WMMA helpers (CDNA5: D = A(16x32 f16) x B(32x16 f16) + C(16x16 f32))
// ---------------------------------------------------------------------------
__device__ inline f8v wmma16(h16v a, h16v b, f8v c) {
  return __builtin_amdgcn_wmma_f32_16x16x32_f16(false, a, false, b, (short)0, c,
                                                false, false);
}

// A-fragment from row-major f16 LDS buffer (ldk halves per row).
// Contiguous-k order (same as pack_w B): lane<16 holds row=lane, K=0..15;
// lane>=16 holds row=lane-16, K=16..31. Two 16B DS loads per lane.
__device__ inline h16v load_a_frag(const _Float16* buf, int ldk, int ktBase, int lane) {
  int row  = lane & 15;
  int kOff = ktBase + ((lane & 16) ? 16 : 0);
  const _Float16* p = buf + row * ldk + kOff;
  h8v lo = *(const h8v*)(p);
  h8v hi = *(const h8v*)(p + 8);
  h16v r;
#pragma unroll
  for (int i = 0; i < 8; i++) { r[i] = lo[i]; r[8 + i] = hi[i]; }
  return r;
}

// B-fragment from pre-packed weights: 32 lanes x 16 f16 (32B/lane, contiguous).
__device__ inline h16v load_b_frag(const _Float16* __restrict__ P, size_t frag, int lane) {
  return *(const h16v*)(P + (frag * 32 + (size_t)lane) * 16);
}

__device__ inline float sigf_(float x)  { return 1.f / (1.f + expf(-x)); }
__device__ inline float geluf_(float x) { return 0.5f * x * (1.f + erff(x * 0.70710678118654752f)); }

template <int ACT>
__device__ inline float act_apply(float x) {
  if (ACT == 1) return tanhf(x);
  if (ACT == 2) return geluf_(x);
  return x;
}

// Async byte-copy global(f16) -> LDS, 16B per lane-op. ASYNCcnt-tracked DMA;
// caller must s_wait_asynccnt + barrier before any wave reads the LDS data.
__device__ inline void async_copy_b128(const _Float16* g, _Float16* l) {
  unsigned lds = (unsigned)(uintptr_t)l;  // LDS aperture keeps offset in low 32b
  asm volatile("global_load_async_to_lds_b128 %0, %1, off"
               :
               : "v"(lds), "v"(g)
               : "memory");
}
__device__ inline void wait_async0() {
  asm volatile("s_wait_asynccnt 0x0" ::: "memory");
}

// ---------------------------------------------------------------------------
// Weight packer: W is (N,K) row-major f32 (torch layout, used as X @ W^T).
// Packed fragment layout: [kt][nt][lane][16 halves]; lane<16 -> col n=nt*16+lane,
// K=kt*32+0..15 ; lane>=16 -> same col, K=kt*32+16..31 (contiguous f32 reads).
// ---------------------------------------------------------------------------
__global__ void pack_w(const float* __restrict__ W, _Float16* __restrict__ P,
                       int K, int N) {
  int tid  = blockIdx.x * blockDim.x + threadIdx.x;
  int lane = tid & 31;
  int frag = tid >> 5;
  int Ktiles = K >> 5, Ntiles = N >> 4;
  if (frag >= Ktiles * Ntiles) return;
  int nt = frag % Ntiles;
  int kt = frag / Ntiles;
  int n = nt * 16 + (lane & 15);
  int k = kt * 32 + ((lane & 16) ? 16 : 0);
  const float* src = W + (size_t)n * K + k;
  _Float16* dst = P + ((size_t)frag * 32 + lane) * 16;
#pragma unroll
  for (int i = 0; i < 16; i++) dst[i] = (_Float16)src[i];
}

__global__ void fill_zero(float* p, int n) {
  int i = blockIdx.x * blockDim.x + threadIdx.x;
  if (i < n) p[i] = 0.f;
}

__global__ void add2(const float* a, const float* b, float* o, int n) {
  int i = blockIdx.x * blockDim.x + threadIdx.x;
  if (i < n) o[i] = a[i] + b[i];
}

// ---------------------------------------------------------------------------
// fc_in MLP: 4 -> 32 (gelu) -> 64 (gelu) -> 128. One wave per row, 8 rows/block.
// rows is always a multiple of 8. Optional f32 and f16 outputs.
// ---------------------------------------------------------------------------
__global__ __launch_bounds__(256) void fc_in_kernel(
    const float* __restrict__ x, const float* __restrict__ w1, const float* __restrict__ b1,
    const float* __restrict__ w2, const float* __restrict__ b2,
    const float* __restrict__ w3, const float* __restrict__ b3,
    float* __restrict__ out32, _Float16* __restrict__ out16, int rows) {
  __shared__ float h1s[8][32];
  __shared__ float h2s[8][64];
  int wave = threadIdx.x >> 5, lane = threadIdx.x & 31;
  int row = blockIdx.x * 8 + wave;
  float xin[4];
#pragma unroll
  for (int i = 0; i < 4; i++) xin[i] = x[(size_t)row * 4 + i];
  {
    float acc = b1[lane];
#pragma unroll
    for (int i = 0; i < 4; i++) acc += xin[i] * w1[lane * 4 + i];
    h1s[wave][lane] = geluf_(acc);
  }
  __syncthreads();
  for (int j = lane; j < 64; j += 32) {
    float acc = b2[j];
#pragma unroll
    for (int i = 0; i < 32; i++) acc += h1s[wave][i] * w2[j * 32 + i];
    h2s[wave][j] = geluf_(acc);
  }
  __syncthreads();
  for (int j = lane; j < 128; j += 32) {
    float acc = b3[j];
#pragma unroll
    for (int i = 0; i < 64; i++) acc += h2s[wave][i] * w3[j * 64 + i];
    if (out32) out32[(size_t)row * 128 + j] = acc;
    if (out16) out16[(size_t)row * 128 + j] = (_Float16)acc;
  }
}

// ---------------------------------------------------------------------------
// Generic WMMA GEMM: C[M,N] = act(A[M,K] @ packedB + bias) (+ residual)
// grid.x = M/16, 256 threads. K = 1<<kshift (128/512/1024). N <= 256.
// Wave w handles N-tiles {w, w+8} (wave-uniform validity -> EXEC stays full).
// Optional f16 mirror output C16 (for async-consumable activations).
// ---------------------------------------------------------------------------
template <int ACT, bool RES>
__global__ __launch_bounds__(256) void gemm_kernel(
    const float* __restrict__ A, const _Float16* __restrict__ Bp,
    const float* __restrict__ bias, const float* __restrict__ resid,
    float* __restrict__ C, _Float16* __restrict__ C16, int K, int N, int kshift) {
  extern __shared__ _Float16 smem[];  // 16*K halves, A tile converted to f16
  int wave = threadIdx.x >> 5, lane = threadIdx.x & 31;
  int m0 = blockIdx.x * 16;
  int tot = 16 << kshift;
  for (int idx = threadIdx.x; idx < tot; idx += 256) {
    int r = idx >> kshift, k = idx & (K - 1);
    smem[idx] = (_Float16)A[((size_t)(m0 + r) << kshift) + k];
  }
  __syncthreads();
  int Ntiles = N >> 4;
  int nt0 = wave, nt1 = wave + 8;
  bool v0 = nt0 < Ntiles, v1 = nt1 < Ntiles;
  f8v acc0 = {}, acc1 = {};
  if (v0) {
    float bv = bias[nt0 * 16 + (lane & 15)];
#pragma unroll
    for (int c = 0; c < 8; c++) acc0[c] = bv;
  }
  if (v1) {
    float bv = bias[nt1 * 16 + (lane & 15)];
#pragma unroll
    for (int c = 0; c < 8; c++) acc1[c] = bv;
  }
  int Ktiles = K >> 5;
  for (int kt = 0; kt < Ktiles; kt++) {
    h16v a = load_a_frag(smem, K, kt * 32, lane);
    h16v b0 = {}, b1 = {};
    if (v0) b0 = load_b_frag(Bp, (size_t)kt * Ntiles + nt0, lane);
    if (v1) b1 = load_b_frag(Bp, (size_t)kt * Ntiles + nt1, lane);
    if (v0) acc0 = wmma16(a, b0, acc0);
    if (v1) acc1 = wmma16(a, b1, acc1);
  }
  int mo = (lane & 16) ? 8 : 0;
  if (v0) {
    int n = nt0 * 16 + (lane & 15);
#pragma unroll
    for (int r = 0; r < 8; r++) {
      size_t m = (size_t)(m0 + r + mo);
      float v = act_apply<ACT>(acc0[r]);
      if (RES) v += resid[m * N + n];
      C[m * N + n] = v;
      if (C16) C16[m * N + n] = (_Float16)v;
    }
  }
  if (v1) {
    int n = nt1 * 16 + (lane & 15);
#pragma unroll
    for (int r = 0; r < 8; r++) {
      size_t m = (size_t)(m0 + r + mo);
      float v = act_apply<ACT>(acc1[r]);
      if (RES) v += resid[m * N + n];
      C[m * N + n] = v;
      if (C16) C16[m * N + n] = (_Float16)v;
    }
  }
}

// ---------------------------------------------------------------------------
// Fused LSTM layer (one direction). grid = B/16 blocks, 256 threads (8 waves).
// Per block: 16 batch rows. Wave w owns hidden units [w*64, w*64+64) -> the
// 4 gate accumulators for a hidden unit share lane/VGPR positions, so the
// c/h update is pure per-component VALU with c kept in registers.
// Per step: gates = bias + x_t @ Wih^T + h @ Whh^T  (all WMMA, f32 accum).
// x_t arrives via global_load_async_to_lds_b128 (input is a f16 mirror).
// B-fragments are batched 8-wide so ~16 loads are in flight per WMMA group.
// ---------------------------------------------------------------------------
__global__ __launch_bounds__(256) void lstm_kernel(
    const _Float16* __restrict__ X,         // (B, Tlen, Kin) f16 mirror
    const _Float16* __restrict__ Wih,       // packed, K=Kin, N=2048
    const _Float16* __restrict__ Whh,       // packed, K=512, N=2048
    const float* __restrict__ bih, const float* __restrict__ bhh,
    const float* __restrict__ h0, const float* __restrict__ c0,  // (B,512)
    float* __restrict__ HSf,                // optional f32 hidden-seq out
    _Float16* __restrict__ HS16,            // optional f16 hidden-seq out
    float* __restrict__ hT, float* __restrict__ cT,              // (B,512)
    int Tlen, int Kin, int kinShift, int ldh, int coff, int rev) {
  extern __shared__ _Float16 smem[];
  _Float16* xbuf = smem;               // 16*Kin halves
  _Float16* hbuf = smem + 16 * Kin;    // 16*512 halves
  int wave = threadIdx.x >> 5, lane = threadIdx.x & 31;
  int mb = blockIdx.x * 16;
  int KinTiles = Kin >> 5;
  int hu0 = wave * 64;
  int laneN = lane & 15;
  int mofs = (lane & 16) ? 8 : 0;
  int chunksPerRow = Kin >> 3;         // 16B chunks per row

  // initial hidden (f16 in LDS) and cell (f32 in regs)
  for (int idx = threadIdx.x; idx < 16 * 512; idx += 256) {
    int r = idx >> 9, j = idx & 511;
    hbuf[idx] = (_Float16)h0[((size_t)(mb + r) << 9) + j];
  }
  f8v creg[4];
#pragma unroll
  for (int tt = 0; tt < 4; tt++) {
    int n = hu0 + tt * 16 + laneN;
#pragma unroll
    for (int r = 0; r < 8; r++)
      creg[tt][r] = c0[((size_t)(mb + r + mofs) << 9) + n];
  }
  // folded biases per gate tile (torch gate order i,f,g,o along 4H axis)
  float btile[16];
#pragma unroll
  for (int gi = 0; gi < 4; gi++)
#pragma unroll
    for (int tt = 0; tt < 4; tt++) {
      int col = gi * 512 + hu0 + tt * 16 + laneN;
      btile[gi * 4 + tt] = bih[col] + bhh[col];
    }
  __syncthreads();

  for (int tstep = 0; tstep < Tlen; tstep++) {
    int t = rev ? (Tlen - 1 - tstep) : tstep;
    // stage x_t via async DMA (pure byte copy, f16 -> f16)
    int chunks = 16 * chunksPerRow;
    for (int idx = threadIdx.x; idx < chunks; idx += 256) {
      int r = idx / chunksPerRow;
      int ko = (idx - r * chunksPerRow) << 3;
      const _Float16* g = X + (((size_t)(mb + r) * Tlen) + t) * Kin + ko;
      async_copy_b128(g, xbuf + (r << kinShift) + ko);
    }
    wait_async0();
    __syncthreads();

    f8v acc[16];
#pragma unroll
    for (int i = 0; i < 16; i++) {
      f8v a;
#pragma unroll
      for (int c = 0; c < 8; c++) a[c] = btile[i];
      acc[i] = a;
    }
    // input-projection contribution (8-wide batched B-frag loads)
    for (int kt = 0; kt < KinTiles; kt++) {
      h16v a = load_a_frag(xbuf, Kin, kt * 32, lane);
      h16v bf[8];
#pragma unroll
      for (int u = 0; u < 8; u++)
        bf[u] = load_b_frag(Wih, (size_t)kt * 128 + (u >> 2) * 32 + wave * 4 + (u & 3), lane);
#pragma unroll
      for (int u = 0; u < 8; u++) acc[u] = wmma16(a, bf[u], acc[u]);
#pragma unroll
      for (int u = 0; u < 8; u++)
        bf[u] = load_b_frag(Wih, (size_t)kt * 128 + (2 + (u >> 2)) * 32 + wave * 4 + (u & 3), lane);
#pragma unroll
      for (int u = 0; u < 8; u++) acc[8 + u] = wmma16(a, bf[u], acc[8 + u]);
    }
    // recurrent contribution (Whh resident in L2: 2MB f16)
    for (int kt = 0; kt < 16; kt++) {
      if (kt < 15)
        __builtin_prefetch(Whh + ((size_t)(kt + 1) * 128 + wave * 4) * 512, 0, 0);
      h16v a = load_a_frag(hbuf, 512, kt * 32, lane);
      h16v bf[8];
#pragma unroll
      for (int u = 0; u < 8; u++)
        bf[u] = load_b_frag(Whh, (size_t)kt * 128 + (u >> 2) * 32 + wave * 4 + (u & 3), lane);
#pragma unroll
      for (int u = 0; u < 8; u++) acc[u] = wmma16(a, bf[u], acc[u]);
#pragma unroll
      for (int u = 0; u < 8; u++)
        bf[u] = load_b_frag(Whh, (size_t)kt * 128 + (2 + (u >> 2)) * 32 + wave * 4 + (u & 3), lane);
#pragma unroll
      for (int u = 0; u < 8; u++) acc[8 + u] = wmma16(a, bf[u], acc[8 + u]);
    }
    __syncthreads();  // all waves done reading hbuf before it is overwritten

    // gate nonlinearity + state update (same lane/VGPR across the 4 gates)
    bool last = (tstep == Tlen - 1);
#pragma unroll
    for (int tt = 0; tt < 4; tt++) {
      f8v gI = acc[0 * 4 + tt], gF = acc[1 * 4 + tt];
      f8v gG = acc[2 * 4 + tt], gO = acc[3 * 4 + tt];
      int n = hu0 + tt * 16 + laneN;
#pragma unroll
      for (int r = 0; r < 8; r++) {
        float i_ = sigf_(gI[r]), f_ = sigf_(gF[r]);
        float g_ = tanhf(gG[r]), o_ = sigf_(gO[r]);
        float cn = f_ * creg[tt][r] + i_ * g_;
        creg[tt][r] = cn;
        float h_ = o_ * tanhf(cn);
        int m = r + mofs;
        hbuf[(m << 9) + n] = (_Float16)h_;
        size_t orow = (((size_t)(mb + m) * Tlen) + t) * ldh + coff + n;
        if (HSf)  HSf[orow] = h_;
        if (HS16) HS16[orow] = (_Float16)h_;
        if (last) {
          hT[((size_t)(mb + m) << 9) + n] = h_;
          cT[((size_t)(mb + m) << 9) + n] = cn;
        }
      }
    }
    __syncthreads();  // hbuf fully written before next step's A-frag reads
  }
}

// ---------------------------------------------------------------------------
// Per-(batch,head) attention, online softmax. block = 64 threads (1 q-row each).
// Q: (B,T,128); KV: (B,S,256) [k cols 0..127, v cols 128..255]; O: (B,T,128).
// ---------------------------------------------------------------------------
__global__ __launch_bounds__(64) void attn_kernel(const float* __restrict__ Q,
                                                  const float* __restrict__ KV,
                                                  float* __restrict__ O) {
  __shared__ float ks[S_ * 32];
  __shared__ float vs[S_ * 32];
  int b = blockIdx.x >> 2;
  int h = blockIdx.x & 3;
  int tid = threadIdx.x;
  for (int idx = tid; idx < S_ * 32; idx += 64) {
    int s = idx >> 5, d = idx & 31;
    size_t row = ((size_t)b * S_ + s) * 256;
    ks[idx] = KV[row + h * 32 + d];
    vs[idx] = KV[row + 128 + h * 32 + d];
  }
  __syncthreads();
  int r = tid;
  float q[32];
#pragma unroll
  for (int d = 0; d < 32; d++) q[d] = Q[((size_t)b * T_ + r) * 128 + h * 32 + d];
  float m = -3.0e38f, l = 0.f;
  float o[32];
#pragma unroll
  for (int d = 0; d < 32; d++) o[d] = 0.f;
  for (int s = 0; s < S_; s++) {
    float x = 0.f;
#pragma unroll
    for (int d = 0; d < 32; d++) x += q[d] * ks[s * 32 + d];
    x *= 0.17677669529663687f;  // 1/sqrt(32)
    float mn = fmaxf(m, x);
    float corr = expf(m - mn);
    float e = expf(x - mn);
    l = l * corr + e;
#pragma unroll
    for (int d = 0; d < 32; d++) o[d] = o[d] * corr + e * vs[s * 32 + d];
    m = mn;
  }
  float inv = 1.f / l;
#pragma unroll
  for (int d = 0; d < 32; d++)
    O[((size_t)b * T_ + r) * 128 + h * 32 + d] = o[d] * inv;
}

// ---------------------------------------------------------------------------
// fc_out tail: h1(rows,64) -> 32 (gelu) -> 4, + trg -> out. One wave per row.
// ---------------------------------------------------------------------------
__global__ __launch_bounds__(256) void fc_out_tail(
    const float* __restrict__ h1, const float* __restrict__ w2, const float* __restrict__ b2,
    const float* __restrict__ w3, const float* __restrict__ b3,
    const float* __restrict__ trg, float* __restrict__ out, int rows) {
  __shared__ float h2s[8][32];
  int wave = threadIdx.x >> 5, lane = threadIdx.x & 31;
  int row = blockIdx.x * 8 + wave;
  float acc = b2[lane];
#pragma unroll
  for (int i = 0; i < 64; i++) acc += h1[(size_t)row * 64 + i] * w2[lane * 64 + i];
  h2s[wave][lane] = geluf_(acc);
  __syncthreads();
  if (lane < 4) {
    float a = b3[lane];
#pragma unroll
    for (int i = 0; i < 32; i++) a += h2s[wave][i] * w3[lane * 32 + i];
    out[(size_t)row * 4 + lane] = trg[(size_t)row * 4 + lane] + a;
  }
}

// ---------------------------------------------------------------------------
// Host orchestration
// ---------------------------------------------------------------------------
extern "C" void kernel_launch(void* const* d_in, const int* in_sizes, int n_in,
                              void* d_out, int out_size, void* d_ws, size_t ws_size,
                              hipStream_t stream) {
  (void)in_sizes; (void)n_in; (void)out_size; (void)ws_size;
  const float* src     = (const float*)d_in[0];
  const float* trg     = (const float*)d_in[1];
  const float* fi_w1   = (const float*)d_in[2];
  const float* fi_b1   = (const float*)d_in[3];
  const float* fi_w2   = (const float*)d_in[4];
  const float* fi_b2   = (const float*)d_in[5];
  const float* fi_w3   = (const float*)d_in[6];
  const float* fi_b3   = (const float*)d_in[7];
  const float* e0f_wih = (const float*)d_in[8];
  const float* e0f_whh = (const float*)d_in[9];
  const float* e0f_bih = (const float*)d_in[10];
  const float* e0f_bhh = (const float*)d_in[11];
  const float* e0b_wih = (const float*)d_in[12];
  const float* e0b_whh = (const float*)d_in[13];
  const float* e0b_bih = (const float*)d_in[14];
  const float* e0b_bhh = (const float*)d_in[15];
  const float* e1f_wih = (const float*)d_in[16];
  const float* e1f_whh = (const float*)d_in[17];
  const float* e1f_bih = (const float*)d_in[18];
  const float* e1f_bhh = (const float*)d_in[19];
  const float* e1b_wih = (const float*)d_in[20];
  const float* e1b_whh = (const float*)d_in[21];
  const float* e1b_bih = (const float*)d_in[22];
  const float* e1b_bhh = (const float*)d_in[23];
  const float* attnfc_w = (const float*)d_in[24];
  const float* attnfc_b = (const float*)d_in[25];
  const float* inproj_w = (const float*)d_in[26];
  const float* inproj_b = (const float*)d_in[27];
  const float* outproj_w = (const float*)d_in[28];
  const float* outproj_b = (const float*)d_in[29];
  const float* d0_wih = (const float*)d_in[30];
  const float* d0_whh = (const float*)d_in[31];
  const float* d0_bih = (const float*)d_in[32];
  const float* d0_bhh = (const float*)d_in[33];
  const float* d1_wih = (const float*)d_in[34];
  const float* d1_whh = (const float*)d_in[35];
  const float* d1_bih = (const float*)d_in[36];
  const float* d1_bhh = (const float*)d_in[37];
  const float* fo_w1 = (const float*)d_in[38];
  const float* fo_b1 = (const float*)d_in[39];
  const float* fo_w2 = (const float*)d_in[40];
  const float* fo_b2 = (const float*)d_in[41];
  const float* fo_w3 = (const float*)d_in[42];
  const float* fo_b3 = (const float*)d_in[43];
  float* out = (float*)d_out;

  char* base = (char*)d_ws;
  size_t off = 0;
  auto A = [&](size_t bytes) -> void* {
    void* p = base + off;
    off += (bytes + 255) & ~(size_t)255;
    return p;
  };
  const int G = 4 * HID_;  // 2048
  // packed f16 weights
  _Float16* pk_e0f_wih = (_Float16*)A((size_t)128 * G * 2);
  _Float16* pk_e0f_whh = (_Float16*)A((size_t)512 * G * 2);
  _Float16* pk_e0b_wih = (_Float16*)A((size_t)128 * G * 2);
  _Float16* pk_e0b_whh = (_Float16*)A((size_t)512 * G * 2);
  _Float16* pk_e1f_wih = (_Float16*)A((size_t)1024 * G * 2);
  _Float16* pk_e1f_whh = (_Float16*)A((size_t)512 * G * 2);
  _Float16* pk_e1b_wih = (_Float16*)A((size_t)1024 * G * 2);
  _Float16* pk_e1b_whh = (_Float16*)A((size_t)512 * G * 2);
  _Float16* pk_d0_wih  = (_Float16*)A((size_t)128 * G * 2);
  _Float16* pk_d0_whh  = (_Float16*)A((size_t)512 * G * 2);
  _Float16* pk_d1_wih  = (_Float16*)A((size_t)512 * G * 2);
  _Float16* pk_d1_whh  = (_Float16*)A((size_t)512 * G * 2);
  _Float16* pk_attnfc  = (_Float16*)A((size_t)1024 * 128 * 2);
  _Float16* pk_q       = (_Float16*)A((size_t)128 * 128 * 2);
  _Float16* pk_kv      = (_Float16*)A((size_t)128 * 256 * 2);
  _Float16* pk_op      = (_Float16*)A((size_t)128 * 128 * 2);
  _Float16* pk_fo1     = (_Float16*)A((size_t)512 * 64 * 2);
  // states
  size_t stB = (size_t)B_ * 512 * 4;
  float* zeros = (float*)A(stB);
  float* hf0 = (float*)A(stB); float* cf0 = (float*)A(stB);
  float* hb0 = (float*)A(stB); float* cb0 = (float*)A(stB);
  float* hf1 = (float*)A(stB); float* cf1 = (float*)A(stB);
  float* hb1 = (float*)A(stB); float* cb1 = (float*)A(stB);
  float* hl0 = (float*)A(stB); float* cl0 = (float*)A(stB);
  float* hl1 = (float*)A(stB); float* cl1 = (float*)A(stB);
  float* dumH = (float*)A(stB); float* dumC = (float*)A(stB);
  // activations (f16 mirrors feed the async-to-LDS LSTM staging)
  _Float16* enc_embed16 = (_Float16*)A((size_t)B_ * S_ * 128 * 2);
  float*    inp         = (float*)A((size_t)B_ * T_ * 128 * 4);
  _Float16* x1_16       = (_Float16*)A((size_t)B_ * S_ * 1024 * 2);
  float*    enc_out     = (float*)A((size_t)B_ * S_ * 1024 * 4);
  float*    proj_enc    = (float*)A((size_t)B_ * S_ * 128 * 4);
  // decoder-phase buffers overlay enc_out (dead after attnfc projection)
  {
    size_t roff = 0;
    char* rb = (char*)enc_out;
    auto R = [&](size_t bytes) -> void* {
      void* p = rb + roff;
      roff += (bytes + 255) & ~(size_t)255;
      return p;
    };
    float*    qbuf     = (float*)R((size_t)B_ * T_ * 128 * 4);
    float*    kvbuf    = (float*)R((size_t)B_ * S_ * 256 * 4);
    float*    obuf     = (float*)R((size_t)B_ * T_ * 128 * 4);
    float*    dec_in   = (float*)R((size_t)B_ * T_ * 128 * 4);
    _Float16* dec_in16 = (_Float16*)R((size_t)B_ * T_ * 128 * 2);
    _Float16* d0out16  = (_Float16*)R((size_t)B_ * T_ * 512 * 2);
    float*    d1out    = (float*)R((size_t)B_ * T_ * 512 * 4);
    float*    h1buf    = (float*)R((size_t)B_ * T_ * 64 * 4);

    auto pack = [&](const float* W, _Float16* P, int K, int N) {
      int threads = (K / 32) * (N / 16) * 32;
      pack_w<<<(threads + 255) / 256, 256, 0, stream>>>(W, P, K, N);
    };
    pack(e0f_wih, pk_e0f_wih, 128, G);  pack(e0f_whh, pk_e0f_whh, 512, G);
    pack(e0b_wih, pk_e0b_wih, 128, G);  pack(e0b_whh, pk_e0b_whh, 512, G);
    pack(e1f_wih, pk_e1f_wih, 1024, G); pack(e1f_whh, pk_e1f_whh, 512, G);
    pack(e1b_wih, pk_e1b_wih, 1024, G); pack(e1b_whh, pk_e1b_whh, 512, G);
    pack(d0_wih,  pk_d0_wih,  128, G);  pack(d0_whh,  pk_d0_whh,  512, G);
    pack(d1_wih,  pk_d1_wih,  512, G);  pack(d1_whh,  pk_d1_whh,  512, G);
    pack(attnfc_w, pk_attnfc, 1024, 128);
    pack(inproj_w, pk_q, 128, 128);
    pack(inproj_w + (size_t)128 * 128, pk_kv, 128, 256);
    pack(outproj_w, pk_op, 128, 128);
    pack(fo_w1, pk_fo1, 512, 64);

    fill_zero<<<(B_ * 512 + 255) / 256, 256, 0, stream>>>(zeros, B_ * 512);

    // embeddings: encoder needs only the f16 mirror; decoder needs f32 (Q + residual)
    fc_in_kernel<<<(B_ * S_) / 8, 256, 0, stream>>>(src, fi_w1, fi_b1, fi_w2, fi_b2,
                                                    fi_w3, fi_b3, nullptr, enc_embed16,
                                                    B_ * S_);
    fc_in_kernel<<<(B_ * T_) / 8, 256, 0, stream>>>(trg, fi_w1, fi_b1, fi_w2, fi_b2,
                                                    fi_w3, fi_b3, inp, nullptr, B_ * T_);

    auto lstm = [&](const _Float16* X, const _Float16* Wih, const _Float16* Whh,
                    const float* bih, const float* bhh, const float* h0,
                    const float* c0, float* HSf, _Float16* HS16, int ldh, int coff,
                    float* hT, float* cT, int Tlen, int Kin, int ksh, int rev) {
      size_t smem = (size_t)(16 * Kin + 16 * 512) * sizeof(_Float16);
      lstm_kernel<<<B_ / 16, 256, smem, stream>>>(X, Wih, Whh, bih, bhh, h0, c0,
                                                  HSf, HS16, hT, cT, Tlen, Kin, ksh,
                                                  ldh, coff, rev);
    };
    // encoder layer 0 (bidirectional) -> x1 (f16 concat)
    lstm(enc_embed16, pk_e0f_wih, pk_e0f_whh, e0f_bih, e0f_bhh, zeros, zeros,
         nullptr, x1_16, 1024, 0, hf0, cf0, S_, 128, 7, 0);
    lstm(enc_embed16, pk_e0b_wih, pk_e0b_whh, e0b_bih, e0b_bhh, zeros, zeros,
         nullptr, x1_16, 1024, 512, hb0, cb0, S_, 128, 7, 1);
    // encoder layer 1 (bidirectional) -> enc_out (f32 concat, feeds attnfc GEMM)
    lstm(x1_16, pk_e1f_wih, pk_e1f_whh, e1f_bih, e1f_bhh, zeros, zeros,
         enc_out, nullptr, 1024, 0, hf1, cf1, S_, 1024, 10, 0);
    lstm(x1_16, pk_e1b_wih, pk_e1b_whh, e1b_bih, e1b_bhh, zeros, zeros,
         enc_out, nullptr, 1024, 512, hb1, cb1, S_, 1024, 10, 1);
    // sum fwd+bwd final states
    int ns = B_ * 512, gb = (ns + 255) / 256;
    add2<<<gb, 256, 0, stream>>>(hf0, hb0, hl0, ns);
    add2<<<gb, 256, 0, stream>>>(cf0, cb0, cl0, ns);
    add2<<<gb, 256, 0, stream>>>(hf1, hb1, hl1, ns);
    add2<<<gb, 256, 0, stream>>>(cf1, cb1, cl1, ns);
    // proj_enc = tanh(enc_out @ attnfc^T + b)
    gemm_kernel<1, false><<<(B_ * S_) / 16, 256, (size_t)16 * 1024 * 2, stream>>>(
        enc_out, pk_attnfc, attnfc_b, nullptr, proj_enc, nullptr, 1024, 128, 10);
    // Q / KV projections
    gemm_kernel<0, false><<<(B_ * T_) / 16, 256, (size_t)16 * 128 * 2, stream>>>(
        inp, pk_q, inproj_b, nullptr, qbuf, nullptr, 128, 128, 7);
    gemm_kernel<0, false><<<(B_ * S_) / 16, 256, (size_t)16 * 128 * 2, stream>>>(
        proj_enc, pk_kv, inproj_b + 128, nullptr, kvbuf, nullptr, 128, 256, 7);
    // attention
    attn_kernel<<<B_ * NH_, 64, 0, stream>>>(qbuf, kvbuf, obuf);
    // out projection + residual (dec_input = inp + attn_out), f16 mirror for d0
    gemm_kernel<0, true><<<(B_ * T_) / 16, 256, (size_t)16 * 128 * 2, stream>>>(
        obuf, pk_op, outproj_b, inp, dec_in, dec_in16, 128, 128, 7);
    // decoder LSTMs
    lstm(dec_in16, pk_d0_wih, pk_d0_whh, d0_bih, d0_bhh, hl0, cl0,
         nullptr, d0out16, 512, 0, dumH, dumC, T_, 128, 7, 0);
    lstm(d0out16, pk_d1_wih, pk_d1_whh, d1_bih, d1_bhh, hl1, cl1,
         d1out, nullptr, 512, 0, dumH, dumC, T_, 512, 9, 0);
    // fc_out: 512 -> 64 (gelu, WMMA) -> 32 (gelu) -> 4, + trg
    gemm_kernel<2, false><<<(B_ * T_) / 16, 256, (size_t)16 * 512 * 2, stream>>>(
        d1out, pk_fo1, fo_b1, nullptr, h1buf, nullptr, 512, 64, 9);
    fc_out_tail<<<(B_ * T_) / 8, 256, 0, stream>>>(h1buf, fo_w2, fo_b2, fo_w3,
                                                   fo_b3, trg, out, B_ * T_);
  }
}
